// Head_50706383897331
// MI455X (gfx1250) — matrix operand order repaired and checked by reading, not compiled, not run on hardware
//
#include <hip/hip_runtime.h>
#include <hip/hip_bf16.h>
#include <stdint.h>

// ---- CDNA5 WMMA types ----
typedef __attribute__((ext_vector_type(16))) _Float16 v16h;
typedef __attribute__((ext_vector_type(8)))  _Float16 v8h;
typedef __attribute__((ext_vector_type(8)))  float    v8f;
typedef __attribute__((ext_vector_type(4)))  float    f32x4;
typedef __attribute__((ext_vector_type(4)))  unsigned int u32x4;
typedef __attribute__((ext_vector_type(4)))  int      i32x4;

#define N_EMBD 1024
#define HS 64
#define T_SEQ 2048
#define BATCH 8
#define NKS (N_EMBD / 32)

#define AS1 __attribute__((address_space(1)))
#define AS3 __attribute__((address_space(3)))

#if defined(__gfx1250__) && __has_builtin(__builtin_amdgcn_global_load_async_to_lds_b128)
#define HAVE_ASYNC_LDS 1
#else
#define HAVE_ASYNC_LDS 0
#endif

static __device__ inline void wait_async0() {
#if __has_builtin(__builtin_amdgcn_s_wait_asynccnt)
  __builtin_amdgcn_s_wait_asynccnt(0);
#else
  asm volatile("s_wait_asynccnt 0x0" ::: "memory");
#endif
}

#if HAVE_ASYNC_LDS
// 64 bytes per lane: global -> LDS, 4x async b128 (ASYNCcnt path).
static __device__ inline void stage_async64(const float* g, float* l) {
  AS1 i32x4* gp = (AS1 i32x4*)(uintptr_t)g;
  AS3 i32x4* lp = (AS3 i32x4*)(unsigned int)(uintptr_t)l;
  __builtin_amdgcn_global_load_async_to_lds_b128(gp, lp, 0, 0);
  __builtin_amdgcn_global_load_async_to_lds_b128(gp, lp, 16, 0);
  __builtin_amdgcn_global_load_async_to_lds_b128(gp, lp, 32, 0);
  __builtin_amdgcn_global_load_async_to_lds_b128(gp, lp, 48, 0);
}
#endif

static __device__ inline v16h mk16(u32x4 lo, u32x4 hi) {
  v16h r;
  ((u32x4*)&r)[0] = lo;
  ((u32x4*)&r)[1] = hi;
  return r;
}

// A-fragment 16 halves from 16 f32 in LDS: elems 0..7 <- p0[0..7] (K=base..+7),
// elems 8..15 <- p1[0..7] (K=base+8..+15)
static __device__ inline v16h cvtA(const float* p0, const float* p1) {
  f32x4 a = *(const f32x4*)(p0);
  f32x4 b = *(const f32x4*)(p0 + 4);
  f32x4 c = *(const f32x4*)(p1);
  f32x4 d = *(const f32x4*)(p1 + 4);
  v16h r;
#pragma unroll
  for (int i = 0; i < 4; ++i) {
    r[i]      = (_Float16)a[i];
    r[4 + i]  = (_Float16)b[i];
    r[8 + i]  = (_Float16)c[i];
    r[12 + i] = (_Float16)d[i];
  }
  return r;
}

static __device__ inline v8f wmma_f16(v16h a, v16h b, v8f c) {
  return __builtin_amdgcn_wmma_f32_16x16x32_f16(false, a, false, b, (short)0, c,
                                                false, false);
}

static __device__ inline float redmax16(float v) {
#pragma unroll
  for (int off = 8; off; off >>= 1) v = fmaxf(v, __shfl_xor(v, off, 32));
  return v;
}
static __device__ inline float redsum16(float v) {
#pragma unroll
  for (int off = 8; off; off >>= 1) v += __shfl_xor(v, off, 32);
  return v;
}

static __device__ inline void wavefence() {
  __builtin_amdgcn_fence(__ATOMIC_ACQ_REL, "wavefront");
  __builtin_amdgcn_wave_barrier();
}

// -------------------------------------------------------------------------
// Kernel 0: repack Wq/Wk/Wv (fp32 [1024][64]) into f16 B-fragment layout.
// -------------------------------------------------------------------------
__global__ void pack_w(const float* __restrict__ wq, const float* __restrict__ wk,
                       const float* __restrict__ wv, _Float16* __restrict__ wpack) {
  const int f = blockIdx.x;        // 0..383
  const int lane = threadIdx.x;    // 0..31
  const int nt = f & 3;
  const int ks = (f >> 2) & 31;
  const int m = f >> 7;
  const float* w = (m == 0) ? wq : ((m == 1) ? wk : wv);
  const int n = nt * 16 + (lane & 15);
  const int kb = ks * 32 + (lane >> 4) * 16;
  _Float16* dst = wpack + ((long)f * 32 + lane) * 16;
#pragma unroll
  for (int e = 0; e < 16; ++e) dst[e] = (_Float16)w[(long)(kb + e) * HS + n];
}

// -------------------------------------------------------------------------
// Kernel 1: QKV projection, f16 WMMA / f32 accumulate, async LDS staging.
// -------------------------------------------------------------------------
__global__ void qkv_kernel(const float* __restrict__ x,
                           const _Float16* __restrict__ wpack,
                           _Float16* __restrict__ qh, _Float16* __restrict__ kh,
                           _Float16* __restrict__ vt) {
  __shared__ __align__(16) float xs[2][128 * 36];

  const int tid = threadIdx.x;
  const int lane = tid & 31;
  const int wave = tid >> 5;
  const int lo16 = lane & 15;
  const int hi = lane >> 4;
  const long row0 = (long)blockIdx.x * 128;

  const int rr = tid >> 1, hh = tid & 1;
  const float* gsrc = x + (row0 + rr) * N_EMBD + hh * 16;  // + ks*32 per step
  float* l0 = &xs[0][rr * 36 + hh * 16];
  float* l1 = &xs[1][rr * 36 + hh * 16];

  v8f acc[12];
#pragma unroll
  for (int i = 0; i < 12; ++i)
#pragma unroll
    for (int r = 0; r < 8; ++r) acc[i][r] = 0.0f;

  const u32x4* wl = (const u32x4*)wpack + lane * 2;
  auto foff = [](int ks, int i) -> int {   // u32x4 units
    const int m = i >> 2, nt = i & 3;
    return ((m * 32 + ks) * 4 + nt) * 64;
  };

#if HAVE_ASYNC_LDS
  stage_async64(gsrc, l0);
#else
  {
    f32x4 t0 = *(const f32x4*)(gsrc);
    f32x4 t1 = *(const f32x4*)(gsrc + 4);
    f32x4 t2 = *(const f32x4*)(gsrc + 8);
    f32x4 t3 = *(const f32x4*)(gsrc + 12);
    *(f32x4*)(l0) = t0; *(f32x4*)(l0 + 4) = t1;
    *(f32x4*)(l0 + 8) = t2; *(f32x4*)(l0 + 12) = t3;
  }
#endif
  v16h bA = mk16(wl[foff(0, 0)], wl[foff(0, 0) + 1]);
  v16h bB = mk16(wl[foff(0, 1)], wl[foff(0, 1) + 1]);
#if HAVE_ASYNC_LDS
  wait_async0();
#endif
  __syncthreads();

  for (int ks = 0; ks < NKS; ++ks) {
    const int buf = ks & 1;
    float* lnext = buf ? l0 : l1;
#if HAVE_ASYNC_LDS
    if (ks + 1 < NKS) stage_async64(gsrc + (ks + 1) * 32, lnext);
#else
    f32x4 t0{}, t1{}, t2{}, t3{};
    if (ks + 1 < NKS) {
      const float* s2 = gsrc + (ks + 1) * 32;
      t0 = *(const f32x4*)(s2);
      t1 = *(const f32x4*)(s2 + 4);
      t2 = *(const f32x4*)(s2 + 8);
      t3 = *(const f32x4*)(s2 + 12);
    }
#endif
    const float* pr = &xs[buf][(wave * 16 + lo16) * 36 + hi * 16];
    const v16h afrag = cvtA(pr, pr + 8);

#pragma unroll
    for (int i = 0; i < 12; ++i) {
      const v16h bcur = bA;
      bA = bB;
      int ks2 = ks, i2 = i + 2;               // 2-deep fragment pipeline
      if (i2 >= 12) { i2 -= 12; ks2 = (ks + 1 < NKS) ? ks + 1 : 0; }
      const int nb = foff(ks2, i2);
      bB = mk16(wl[nb], wl[nb + 1]);
      acc[i] = wmma_f16(afrag, bcur, acc[i]);
    }

#if HAVE_ASYNC_LDS
    if (ks + 1 < NKS) wait_async0();
#else
    if (ks + 1 < NKS) {
      *(f32x4*)(lnext) = t0; *(f32x4*)(lnext + 4) = t1;
      *(f32x4*)(lnext + 8) = t2; *(f32x4*)(lnext + 12) = t3;
    }
#endif
    __syncthreads();
  }

  const long rbase = row0 + wave * 16 + hi * 8;
  const long bb = rbase >> 11;
  const long tt = rbase & 2047;
#pragma unroll
  for (int nt = 0; nt < 4; ++nt) {
    const int col = nt * 16 + lo16;
#pragma unroll
    for (int r = 0; r < 8; ++r) {
      qh[(rbase + r) * HS + col] = (_Float16)acc[0 * 4 + nt][r];
      kh[(rbase + r) * HS + col] = (_Float16)acc[1 * 4 + nt][r];
    }
    v8h vv;
#pragma unroll
    for (int r = 0; r < 8; ++r) vv[r] = (_Float16)acc[2 * 4 + nt][r];
    *(u32x4*)(vt + (bb * HS + col) * T_SEQ + tt) = *(u32x4*)&vv;
  }
}

// -------------------------------------------------------------------------
// Kernel 2: causal flash attention, K/V fragment loads software-pipelined
// one key-block ahead so the softmax VALU stretch hides load latency.
// -------------------------------------------------------------------------
__global__ void attn_kernel(const _Float16* __restrict__ qh,
                            const _Float16* __restrict__ kh,
                            const _Float16* __restrict__ vt,
                            float* __restrict__ out) {
  __shared__ __align__(16) _Float16 ps[8][16 * 40];  // per-wave P buffer

  const int lane = threadIdx.x & 31;
  const int wave = threadIdx.x >> 5;
  const int lo16 = lane & 15;
  const int hi = lane >> 4;
  const int b = blockIdx.x;
  const int q0 = blockIdx.y * 128 + wave * 16;
  const long qrow = (long)b * T_SEQ + q0;

  v16h qf[2];
  {
    const u32x4* q4 = (const u32x4*)(qh + (qrow + lo16) * HS);
#pragma unroll
    for (int kb = 0; kb < 2; ++kb) qf[kb] = mk16(q4[kb * 4 + hi * 2], q4[kb * 4 + hi * 2 + 1]);
  }

  const u32x4* kp = (const u32x4*)(kh + (long)b * T_SEQ * HS);
  const u32x4* vp = (const u32x4*)vt;

  auto loadblock = [&](int jb, v16h (&kf)[2][2], v16h (&vf)[4]) {
#pragma unroll
    for (int jt = 0; jt < 2; ++jt) {
      const long key = jb + jt * 16 + lo16;
#pragma unroll
      for (int kb = 0; kb < 2; ++kb) {
        const long kidx = key * 8 + kb * 4 + hi * 2;
        kf[jt][kb] = mk16(kp[kidx], kp[kidx + 1]);
      }
    }
#pragma unroll
    for (int nt = 0; nt < 4; ++nt) {
      const long vrow = ((long)b * HS + nt * 16 + lo16) * (T_SEQ / 8) + (jb >> 3) + hi * 2;
      vf[nt] = mk16(vp[vrow], vp[vrow + 1]);
    }
  };

  float m[8], l[8];
  v8f o[4];
#pragma unroll
  for (int r = 0; r < 8; ++r) { m[r] = -__builtin_inff(); l[r] = 0.0f; }
#pragma unroll
  for (int nt = 0; nt < 4; ++nt)
#pragma unroll
    for (int r = 0; r < 8; ++r) o[nt][r] = 0.0f;

  const float scale = 0.03125f;  // n_embd^-0.5

  v16h kfc[2][2], vfc[4];
  v16h kfn[2][2], vfn[4];
  loadblock(0, kfc, vfc);

  for (int j0 = 0; j0 < q0 + 16; j0 += 32) {
    // issue next block's loads NOW; their wait lands after softmax + 4 WMMAs
    const int jn = (j0 + 32 < q0 + 16) ? (j0 + 32) : j0;
    loadblock(jn, kfn, vfn);

    // ---- S = scale * Q K^T ----
    v8f s[2];
#pragma unroll
    for (int jt = 0; jt < 2; ++jt) {
      v8f sa;
#pragma unroll
      for (int r = 0; r < 8; ++r) sa[r] = 0.0f;
#pragma unroll
      for (int kb = 0; kb < 2; ++kb) sa = wmma_f16(qf[kb], kfc[jt][kb], sa);
      s[jt] = sa;
    }
    // ---- scale + causal mask ----
#pragma unroll
    for (int jt = 0; jt < 2; ++jt) {
      const int key = j0 + jt * 16 + lo16;
#pragma unroll
      for (int r = 0; r < 8; ++r) {
        const int row = q0 + r + hi * 8;
        const float v = s[jt][r] * scale;
        s[jt][r] = (key <= row) ? v : -__builtin_inff();
      }
    }
    // ---- online softmax ----
    float alpha[8];
#pragma unroll
    for (int r = 0; r < 8; ++r) {
      float mx = redmax16(fmaxf(s[0][r], s[1][r]));
      const float mn = fmaxf(m[r], mx);
      const float p0 = __expf(s[0][r] - mn);
      const float p1 = __expf(s[1][r] - mn);
      s[0][r] = p0;
      s[1][r] = p1;
      const float rs = redsum16(p0 + p1);
      alpha[r] = __expf(m[r] - mn);
      l[r] = l[r] * alpha[r] + rs;
      m[r] = mn;
    }
#pragma unroll
    for (int nt = 0; nt < 4; ++nt)
#pragma unroll
      for (int r = 0; r < 8; ++r) o[nt][r] *= alpha[r];

    // ---- P: C-layout -> A-layout via wave-local LDS ----
#pragma unroll
    for (int jt = 0; jt < 2; ++jt)
#pragma unroll
      for (int r = 0; r < 8; ++r)
        ps[wave][(r + hi * 8) * 40 + jt * 16 + lo16] = (_Float16)s[jt][r];
    wavefence();
    const u32x4* pp = (const u32x4*)&ps[wave][0];
    const v16h pa = mk16(pp[lo16 * 5 + hi * 2], pp[lo16 * 5 + hi * 2 + 1]);
    wavefence();

    // ---- O += P V ----
#pragma unroll
    for (int nt = 0; nt < 4; ++nt) o[nt] = wmma_f16(pa, vfc[nt], o[nt]);

    // rotate pipeline
#pragma unroll
    for (int jt = 0; jt < 2; ++jt)
#pragma unroll
      for (int kb = 0; kb < 2; ++kb) kfc[jt][kb] = kfn[jt][kb];
#pragma unroll
    for (int nt = 0; nt < 4; ++nt) vfc[nt] = vfn[nt];
  }

  // ---- epilogue ----
  float inv[8];
#pragma unroll
  for (int r = 0; r < 8; ++r) inv[r] = 1.0f / l[r];
#pragma unroll
  for (int nt = 0; nt < 4; ++nt) {
    const int col = nt * 16 + lo16;
#pragma unroll
    for (int r = 0; r < 8; ++r)
      out[(qrow + r + hi * 8) * HS + col] = o[nt][r] * inv[r];
  }
}

// -------------------------------------------------------------------------
extern "C" void kernel_launch(void* const* d_in, const int* in_sizes, int n_in,
                              void* d_out, int out_size, void* d_ws, size_t ws_size,
                              hipStream_t stream) {
  const float* x = (const float*)d_in[0];
  const float* wq = (const float*)d_in[1];
  const float* wk = (const float*)d_in[2];
  const float* wv = (const float*)d_in[3];
  float* out = (float*)d_out;

  // workspace (f16): wpack 384 KiB | Qh 2 MiB | Kh 2 MiB | Vt 2 MiB
  _Float16* wpack = (_Float16*)d_ws;
  _Float16* qh = (_Float16*)((char*)d_ws + 393216);
  _Float16* kh = qh + (long)BATCH * T_SEQ * HS;
  _Float16* vt = kh + (long)BATCH * T_SEQ * HS;

  pack_w<<<dim3(384), dim3(32), 0, stream>>>(wq, wk, wv, wpack);
  qkv_kernel<<<dim3((BATCH * T_SEQ) / 128), dim3(256), 0, stream>>>(x, wpack, qh, kh, vt);
  attn_kernel<<<dim3(BATCH, T_SEQ / 128), dim3(256), 0, stream>>>(qh, kh, vt, out);
}
